// GLiNERSpanPooler_77927886619145
// MI455X (gfx1250) — matrix operand order repaired and checked by reading, not compiled, not run on hardware
//
#include <hip/hip_runtime.h>
#include <hip/hip_bf16.h>
#include <math.h>

// ---------------------------------------------------------------------------
// GLiNER span pooler for gfx1250 (MI455X): all GEMMs via v_wmma_f32_16x16x32_bf16
// B=8, L=512, H=768, W=12, C=25, S=L*W=6144, Hh=384
// 32 rows per workgroup (2 M-tiles/wave, dual accumulators) so each B fragment
// feeds 2 wmmas -> halves L2 weight traffic and per-wmma memory issue.
// ---------------------------------------------------------------------------

#define Bb   8
#define Ls   512
#define Hs   768
#define Ws   12
#define Cs   25
#define Ss   (Ls * Ws)          // 6144
#define Hh   (Hs / 2)           // 384
#define G4   (4 * Hh)           // 1536
#define H4   (4 * Hs)           // 3072
#define MTOK (Bb * Ls)          // 4096 token rows
#define MSPN (Bb * Ss)          // 49152 span rows
#define MPRM (Bb * Cs)          // 200 prompt rows
#define SPAN_ELEMS ((size_t)Bb * Ls * Ws * Hs)  // 37748736
#define ROWS 32                  // rows per workgroup (2 x 16-row M-tiles)

typedef __attribute__((ext_vector_type(16))) __bf16 v16bf;
typedef __attribute__((ext_vector_type(8)))  float  v8f;

// ---------------- WMMA fragment loaders ------------------------------------
// A (16x32 bf16), 16-bit A layout: lanes 0-15 => M=0..15, K in {0..7,16..23};
// lanes 16-31 => same M, K in {8..15,24..31}. Two contiguous 16B runs per lane.
__device__ inline v16bf load_a_frag(const __bf16* tile, int lda, int kk, int lane) {
  const int m  = lane & 15;
  const int hi = lane >> 4;
  const __bf16* r = tile + (size_t)m * lda + kk;
  const int b0 = hi ? 8 : 0;
  const int b1 = hi ? 24 : 16;
  v16bf a;
#pragma unroll
  for (int e = 0; e < 8; ++e) { a[e] = r[b0 + e]; a[8 + e] = r[b1 + e]; }
  return a;
}

// B (32x16 bf16) from a [K,N] bf16 weight: lanes 0-15 => N=0..15, K in {0..7,16..23};
// lanes 16-31 => same N, K in {8..15,24..31} (matches A layout).
__device__ inline v16bf load_b_frag(const __bf16* wt, int ldb, int kk, int n0, int lane) {
  const int hi = lane >> 4;
  const __bf16* r = wt + (size_t)kk * ldb + n0 + (lane & 15);
  v16bf b;
#pragma unroll
  for (int e = 0; e < 8; ++e) { b[e] = r[(size_t)(8 * hi + e) * ldb]; b[8 + e] = r[(size_t)(16 + 8 * hi + e) * ldb]; }
  return b;
}

// ---------------- weight transpose + cast: w[N,K] f32 -> wt[K,N] bf16 ------
__global__ __launch_bounds__(256) void convT_kernel(const float* __restrict__ w,
                                                    __bf16* __restrict__ wt,
                                                    int N, int K) {
  int i = blockIdx.x * 256 + threadIdx.x;
  if (i < N * K) {
    int n = i / K, k = i - n * K;
    wt[(size_t)k * N + n] = (__bf16)w[i];
  }
}

// ---------------- single GEMM: out[M,N] = A[M,K] @ wt[K,N] + ba + bb -------
__global__ __launch_bounds__(256) void gemm_bias16_kernel(
    int M, int K, int N,
    const float* __restrict__ A32,
    const __bf16* __restrict__ wt,
    const float* __restrict__ bias_a,
    const float* __restrict__ bias_b,
    float* __restrict__ out) {
  extern __shared__ char smem[];
  __bf16* sA = (__bf16*)smem;                 // ROWS*K bf16
  const int tid = threadIdx.x, lane = tid & 31, wave = tid >> 5;
  const int row0 = blockIdx.x * ROWS;

  for (int idx = tid; idx < ROWS * K; idx += 256) {
    int m = idx / K, k = idx - m * K;
    int r = row0 + m;
    sA[idx] = (__bf16)((r < M) ? A32[(size_t)r * K + k] : 0.f);
  }
  __syncthreads();

  const int ntiles = N >> 4;
  for (int nt = wave; nt < ntiles; nt += 8) {
    v8f acc0 = {0.f, 0.f, 0.f, 0.f, 0.f, 0.f, 0.f, 0.f};
    v8f acc1 = {0.f, 0.f, 0.f, 0.f, 0.f, 0.f, 0.f, 0.f};
    const int n0 = nt << 4;
    for (int kk = 0; kk < K; kk += 32) {
      v16bf b  = load_b_frag(wt, N, kk, n0, lane);
      v16bf a0 = load_a_frag(sA,              K, kk, lane);
      v16bf a1 = load_a_frag(sA + 16 * K,     K, kk, lane);
      acc0 = __builtin_amdgcn_wmma_f32_16x16x32_bf16(false, a0, false, b,
                                                     (short)0, acc0, false, false);
      acc1 = __builtin_amdgcn_wmma_f32_16x16x32_bf16(false, a1, false, b,
                                                     (short)0, acc1, false, false);
    }
    const int n  = n0 + (lane & 15);
    const int mb = (lane < 16) ? 0 : 8;
    const float bias = bias_a[n] + bias_b[n];
#pragma unroll
    for (int v = 0; v < 8; ++v) {
      int r0 = row0 + v + mb;
      int r1 = r0 + 16;
      if (r0 < M) out[(size_t)r0 * N + n] = acc0[v] + bias;
      if (r1 < M) out[(size_t)r1 * N + n] = acc1[v] + bias;
    }
  }
}

// ---------------- BiLSTM recurrent scan (one WG per dir x batch) -----------
__global__ __launch_bounds__(512) void lstm_scan_kernel(
    const float* __restrict__ xg_f, const float* __restrict__ xg_b,
    const float* __restrict__ whh_f, const float* __restrict__ whh_b,
    const int* __restrict__ mask, float* __restrict__ hout) {
  __shared__ float sh[Hh], sc[Hh], sg[G4];
  const int dir = blockIdx.x >> 3;
  const int b   = blockIdx.x & 7;
  const float* xg  = dir ? xg_b : xg_f;    // biases already folded into xg
  const float* whh = dir ? whh_b : whh_f;  // [G4, Hh] f32 row-major
  const int tid = threadIdx.x;

  for (int i = tid; i < Hh; i += 512) { sh[i] = 0.f; sc[i] = 0.f; }
  __syncthreads();

  for (int step = 0; step < Ls; ++step) {
    const int t = dir ? (Ls - 1 - step) : step;
    const float* xrow = xg + ((size_t)b * Ls + t) * G4;
    for (int g = tid; g < G4; g += 512) {
      const float4* w4 = (const float4*)(whh + (size_t)g * Hh);
      const float4* h4 = (const float4*)sh;
      float acc = xrow[g];
#pragma unroll 4
      for (int k = 0; k < Hh / 4; ++k) {
        float4 w = w4[k];
        float4 h = h4[k];
        acc += h.x * w.x + h.y * w.y + h.z * w.z + h.w * w.w;
      }
      sg[g] = acc;
    }
    __syncthreads();
    if (tid < Hh) {
      float ig = 1.f / (1.f + __expf(-sg[tid]));
      float fg = 1.f / (1.f + __expf(-sg[Hh + tid]));
      float gg = tanhf(sg[2 * Hh + tid]);
      float og = 1.f / (1.f + __expf(-sg[3 * Hh + tid]));
      float c = fg * sc[tid] + ig * gg;
      float h = og * tanhf(c);
      sc[tid] = c;
      sh[tid] = h;
      const float mk = (float)mask[b * Ls + t];
      hout[((size_t)b * Ls + t) * Hs + dir * Hh + tid] = h * mk;
    }
    __syncthreads();
  }
}

// ---------------- fused 2-stage MLP (ROWS rows / WG), hidden kept in LDS ---
// mode 0: A=f32 dense, output = bf16 relu (token start/end reps)
// mode 1: A gathered from bf16 reps via span_idx, output = f32 (span)
// mode 2: A=f32 dense, output = f32 (prompt)
__global__ __launch_bounds__(256) void mlp16_kernel(
    int mode, int M, int K1, int N1, int N2,
    const float* __restrict__ A32,
    const __bf16* __restrict__ repS, const __bf16* __restrict__ repE,
    const int* __restrict__ spanIdx,
    const __bf16* __restrict__ wt1, const float* __restrict__ b1,
    const __bf16* __restrict__ wt2, const float* __restrict__ b2,
    float* __restrict__ outF, __bf16* __restrict__ outB) {
  extern __shared__ char smem[];
  __bf16* sA = (__bf16*)smem;              // ROWS*K1 bf16
  __bf16* sH = sA + ROWS * K1;             // ROWS*N1 bf16 (relu'd hidden)
  const int tid = threadIdx.x, lane = tid & 31, wave = tid >> 5;
  const int row0 = blockIdx.x * ROWS;

  // ---- stage A tile ----
  if (mode == 1) {
    for (int idx = tid; idx < ROWS * K1; idx += 256) {
      int m = idx / K1, k = idx - m * K1;
      int r = row0 + m;
      __bf16 v = (__bf16)0.f;
      if (r < M) {
        int b = r / Ss, s = r - b * Ss;
        if (k < Hs) {
          int si = spanIdx[((size_t)(b * Ss + s)) * 2 + 0];
          si = si < 0 ? 0 : (si > Ls - 1 ? Ls - 1 : si);
          v = repS[((size_t)(b * Ls + si)) * Hs + k];
        } else {
          int ei = spanIdx[((size_t)(b * Ss + s)) * 2 + 1];
          ei = ei < 0 ? 0 : (ei > Ls - 1 ? Ls - 1 : ei);
          v = repE[((size_t)(b * Ls + ei)) * Hs + (k - Hs)];
        }
      }
      sA[idx] = v;
    }
  } else {
    for (int idx = tid; idx < ROWS * K1; idx += 256) {
      int m = idx / K1, k = idx - m * K1;
      int r = row0 + m;
      sA[idx] = (__bf16)((r < M) ? A32[(size_t)r * K1 + k] : 0.f);
    }
  }
  __syncthreads();

  // ---- GEMM1: [ROWS,K1] @ [K1,N1] -> relu -> bf16 hidden in LDS ----
  const int nt1 = N1 >> 4;
  for (int nt = wave; nt < nt1; nt += 8) {
    v8f acc0 = {0.f, 0.f, 0.f, 0.f, 0.f, 0.f, 0.f, 0.f};
    v8f acc1 = {0.f, 0.f, 0.f, 0.f, 0.f, 0.f, 0.f, 0.f};
    const int n0 = nt << 4;
    for (int kk = 0; kk < K1; kk += 32) {
      v16bf b  = load_b_frag(wt1, N1, kk, n0, lane);
      v16bf a0 = load_a_frag(sA,           K1, kk, lane);
      v16bf a1 = load_a_frag(sA + 16 * K1, K1, kk, lane);
      acc0 = __builtin_amdgcn_wmma_f32_16x16x32_bf16(false, a0, false, b,
                                                     (short)0, acc0, false, false);
      acc1 = __builtin_amdgcn_wmma_f32_16x16x32_bf16(false, a1, false, b,
                                                     (short)0, acc1, false, false);
    }
    const int n  = n0 + (lane & 15);
    const int mb = (lane < 16) ? 0 : 8;
    const float bias = b1[n];
#pragma unroll
    for (int v = 0; v < 8; ++v) {
      float y0 = acc0[v] + bias;
      float y1 = acc1[v] + bias;
      y0 = y0 > 0.f ? y0 : 0.f;
      y1 = y1 > 0.f ? y1 : 0.f;
      sH[(size_t)(v + mb) * N1 + n]        = (__bf16)y0;
      sH[(size_t)(v + mb + 16) * N1 + n]   = (__bf16)y1;
    }
  }
  __syncthreads();

  // ---- GEMM2: [ROWS,N1] @ [N1,N2] + bias -> output ----
  const int nt2 = N2 >> 4;
  for (int nt = wave; nt < nt2; nt += 8) {
    v8f acc0 = {0.f, 0.f, 0.f, 0.f, 0.f, 0.f, 0.f, 0.f};
    v8f acc1 = {0.f, 0.f, 0.f, 0.f, 0.f, 0.f, 0.f, 0.f};
    const int n0 = nt << 4;
    for (int kk = 0; kk < N1; kk += 32) {
      v16bf b  = load_b_frag(wt2, N2, kk, n0, lane);
      v16bf a0 = load_a_frag(sH,           N1, kk, lane);
      v16bf a1 = load_a_frag(sH + 16 * N1, N1, kk, lane);
      acc0 = __builtin_amdgcn_wmma_f32_16x16x32_bf16(false, a0, false, b,
                                                     (short)0, acc0, false, false);
      acc1 = __builtin_amdgcn_wmma_f32_16x16x32_bf16(false, a1, false, b,
                                                     (short)0, acc1, false, false);
    }
    const int n  = n0 + (lane & 15);
    const int mb = (lane < 16) ? 0 : 8;
    const float bias = b2[n];
#pragma unroll
    for (int v = 0; v < 8; ++v) {
      int r0 = row0 + v + mb;
      int r1 = r0 + 16;
      if (r0 < M) {
        float y = acc0[v] + bias;
        if (mode == 0) outB[(size_t)r0 * N2 + n] = (__bf16)(y > 0.f ? y : 0.f);
        else           outF[(size_t)r0 * N2 + n] = y;
      }
      if (r1 < M) {
        float y = acc1[v] + bias;
        if (mode == 0) outB[(size_t)r1 * N2 + n] = (__bf16)(y > 0.f ? y : 0.f);
        else           outF[(size_t)r1 * N2 + n] = y;
      }
    }
  }
}

// ---------------------------------------------------------------------------
extern "C" void kernel_launch(void* const* d_in, const int* in_sizes, int n_in,
                              void* d_out, int out_size, void* d_ws, size_t ws_size,
                              hipStream_t stream) {
  const float* we       = (const float*)d_in[0];
  const int*   mask     = (const int*)  d_in[1];
  const int*   span_idx = (const int*)  d_in[2];
  const float* pe       = (const float*)d_in[3];
  const float* w_ih_f = (const float*)d_in[4];
  const float* w_hh_f = (const float*)d_in[5];
  const float* b_ih_f = (const float*)d_in[6];
  const float* b_hh_f = (const float*)d_in[7];
  const float* w_ih_b = (const float*)d_in[8];
  const float* w_hh_b = (const float*)d_in[9];
  const float* b_ih_b = (const float*)d_in[10];
  const float* b_hh_b = (const float*)d_in[11];
  const float* start_w1 = (const float*)d_in[12]; const float* start_b1 = (const float*)d_in[13];
  const float* start_w2 = (const float*)d_in[14]; const float* start_b2 = (const float*)d_in[15];
  const float* end_w1   = (const float*)d_in[16]; const float* end_b1   = (const float*)d_in[17];
  const float* end_w2   = (const float*)d_in[18]; const float* end_b2   = (const float*)d_in[19];
  const float* out_w1   = (const float*)d_in[20]; const float* out_b1   = (const float*)d_in[21];
  const float* out_w2   = (const float*)d_in[22]; const float* out_b2   = (const float*)d_in[23];
  const float* pr_w1    = (const float*)d_in[24]; const float* pr_b1    = (const float*)d_in[25];
  const float* pr_w2    = (const float*)d_in[26]; const float* pr_b2    = (const float*)d_in[27];

  // ---- workspace carve-out ----
  char* ws = (char*)d_ws;
  size_t off = 0;
  auto carve = [&](size_t bytes) -> char* {
    char* p = ws + off;
    off = (off + bytes + 255) & ~(size_t)255;
    return p;
  };
  __bf16* wtih_f = (__bf16*)carve((size_t)Hs * G4 * 2);
  __bf16* wtih_b = (__bf16*)carve((size_t)Hs * G4 * 2);
  __bf16* sw1t   = (__bf16*)carve((size_t)Hs * H4 * 2);
  __bf16* sw2t   = (__bf16*)carve((size_t)H4 * Hs * 2);
  __bf16* ew1t   = (__bf16*)carve((size_t)Hs * H4 * 2);
  __bf16* ew2t   = (__bf16*)carve((size_t)H4 * Hs * 2);
  __bf16* ow1t   = (__bf16*)carve((size_t)(2 * Hs) * H4 * 2);
  __bf16* ow2t   = (__bf16*)carve((size_t)H4 * Hs * 2);
  __bf16* pw1t   = (__bf16*)carve((size_t)Hs * H4 * 2);
  __bf16* pw2t   = (__bf16*)carve((size_t)H4 * Hs * 2);
  float*  xg_f   = (float*) carve((size_t)MTOK * G4 * 4);
  float*  xg_b   = (float*) carve((size_t)MTOK * G4 * 4);
  float*  hbuf   = (float*) carve((size_t)MTOK * Hs * 4);
  __bf16* srep   = (__bf16*)carve((size_t)MTOK * Hs * 2);
  __bf16* erep   = (__bf16*)carve((size_t)MTOK * Hs * 2);

  (void)hipFuncSetAttribute((const void*)mlp16_kernel,
                            hipFuncAttributeMaxDynamicSharedMemorySize,
                            ROWS * (2 * Hs + H4) * 2);          // 294912 B
  (void)hipFuncSetAttribute((const void*)gemm_bias16_kernel,
                            hipFuncAttributeMaxDynamicSharedMemorySize,
                            ROWS * Hs * 2);                     // 49152 B

  auto convT = [&](const float* w, __bf16* wt, int N, int K) {
    int tot = N * K;
    convT_kernel<<<(tot + 255) / 256, 256, 0, stream>>>(w, wt, N, K);
  };
  // weight casts (transposed to [K,N] bf16)
  convT(w_ih_f,  wtih_f, G4, Hs);
  convT(w_ih_b,  wtih_b, G4, Hs);
  convT(start_w1, sw1t, H4, Hs);
  convT(start_w2, sw2t, Hs, H4);
  convT(end_w1,   ew1t, H4, Hs);
  convT(end_w2,   ew2t, Hs, H4);
  convT(out_w1,   ow1t, H4, 2 * Hs);
  convT(out_w2,   ow2t, Hs, H4);
  convT(pr_w1,    pw1t, H4, Hs);
  convT(pr_w2,    pw2t, Hs, H4);

  // xg = we @ w_ih^T + b_ih + b_hh  (both directions)
  {
    size_t shm = (size_t)ROWS * Hs * 2;
    gemm_bias16_kernel<<<MTOK / ROWS, 256, shm, stream>>>(
        MTOK, Hs, G4, we, wtih_f, b_ih_f, b_hh_f, xg_f);
    gemm_bias16_kernel<<<MTOK / ROWS, 256, shm, stream>>>(
        MTOK, Hs, G4, we, wtih_b, b_ih_b, b_hh_b, xg_b);
  }

  // BiLSTM scan -> hbuf [B,L,768] (masked)
  lstm_scan_kernel<<<16, 512, 0, stream>>>(xg_f, xg_b, w_hh_f, w_hh_b, mask, hbuf);

  // start / end token MLPs -> relu'd bf16 reps
  {
    size_t shm = (size_t)ROWS * (Hs + H4) * 2;  // 245760 B
    mlp16_kernel<<<MTOK / ROWS, 256, shm, stream>>>(
        0, MTOK, Hs, H4, Hs, hbuf, nullptr, nullptr, nullptr,
        sw1t, start_b1, sw2t, start_b2, nullptr, srep);
    mlp16_kernel<<<MTOK / ROWS, 256, shm, stream>>>(
        0, MTOK, Hs, H4, Hs, hbuf, nullptr, nullptr, nullptr,
        ew1t, end_b1, ew2t, end_b2, nullptr, erep);
  }

  // span MLP with fused gather -> d_out[0 .. SPAN_ELEMS)
  {
    size_t shm = (size_t)ROWS * (2 * Hs + H4) * 2;  // 294912 B
    mlp16_kernel<<<MSPN / ROWS, 256, shm, stream>>>(
        1, MSPN, 2 * Hs, H4, Hs, nullptr, srep, erep, span_idx,
        ow1t, out_b1, ow2t, out_b2, (float*)d_out, nullptr);
  }

  // prompt MLP -> d_out[SPAN_ELEMS ..)
  {
    size_t shm = (size_t)ROWS * (Hs + H4) * 2;
    mlp16_kernel<<<(MPRM + ROWS - 1) / ROWS, 256, shm, stream>>>(
        2, MPRM, Hs, H4, Hs, pe, nullptr, nullptr, nullptr,
        pw1t, pr_b1, pw2t, pr_b2, (float*)d_out + SPAN_ELEMS, nullptr);
  }
}
